// Attention_57260503990448
// MI455X (gfx1250) — matrix operand order repaired and checked by reading, not compiled, not run on hardware
//
#include <hip/hip_runtime.h>
#include <hip/hip_bf16.h>

typedef __attribute__((ext_vector_type(16))) _Float16 v16h;
typedef __attribute__((ext_vector_type(8)))  _Float16 v8h;
typedef __attribute__((ext_vector_type(4)))  _Float16 v4h;
typedef __attribute__((ext_vector_type(8)))  float    v8f;
typedef __attribute__((ext_vector_type(4)))  float    v4f;

#define CAT16(lo, hi) __builtin_shufflevector(lo, hi, 0,1,2,3,4,5,6,7,8,9,10,11,12,13,14,15)

#define B_      2
#define S_      2048
#define H_      32
#define HKV_    8
#define D_      128
#define GROUP_  (H_ / HKV_)
#define WINDOW_ 1024
#define SCALE_  0.08838834764831845f   // 1/sqrt(128)
#define CAP_    30.0f

#define QTILE_WAVE 16
#define WAVES      4
#define QTILE_WG   (QTILE_WAVE * WAVES)   // 64
#define KCHUNK     32

#define KT_PITCH 136   // halves (128 + 8 pad) -> 272B rows, 16B aligned
#define VT_PITCH 40    // halves (32 + 8 pad)  -> 80B rows, 16B aligned

__global__ __launch_bounds__(128)
void fa_swa_gqa_kernel(const float* __restrict__ Q,
                       const float* __restrict__ K,
                       const float* __restrict__ V,
                       float* __restrict__ Out)
{
    __shared__ _Float16 kt[KCHUNK * KT_PITCH];   // K chunk  [kv][d]   (f16)
    __shared__ _Float16 vt[D_ * VT_PITCH];       // V chunk  [d][kv]   (f16, transposed)
    __shared__ _Float16 ps[WAVES * 16 * 32];     // per-wave P tile [row][col] (f16)

    const int tid  = threadIdx.x;        // 0..127
    const int wave = tid >> 5;
    const int lane = tid & 31;
    const int half = lane >> 4;          // 0 or 1
    const int l16  = lane & 15;

    const int qtiles = S_ / QTILE_WG;
    const int bx = blockIdx.x;
    const int qt = bx % qtiles;
    const int h  = (bx / qtiles) % H_;
    const int b  = bx / (qtiles * H_);
    const int hkv = h / GROUP_;

    const int q0   = qt * QTILE_WG + wave * QTILE_WAVE; // wave's first query row
    const int qrow = q0 + l16;                          // A-matrix row for this lane

    // ---------------- Q tile -> 4 f16 A-fragments (16x32 each, K = d slice) ----
    // A 16-bit layout: lane row M = lane&15; element e -> K = kd*32 + (e<8 ? 8*half+e
    //                                                              : 16+8*half+(e-8))
    v16h qfrag[4];
    {
        const float* qp = Q + (((size_t)(b * S_ + qrow)) * H_ + h) * D_;
#pragma unroll
        for (int kd = 0; kd < 4; ++kd) {
            const float* p0 = qp + kd * 32 + 8 * half;
            const float* p1 = qp + kd * 32 + 16 + 8 * half;
            v4f a0 = *(const v4f*)(p0);
            v4f a1 = *(const v4f*)(p0 + 4);
            v4f b0 = *(const v4f*)(p1);
            v4f b1 = *(const v4f*)(p1 + 4);
            v16h f;
#pragma unroll
            for (int i = 0; i < 4; ++i) {
                f[i]      = (_Float16)a0[i];
                f[4 + i]  = (_Float16)a1[i];
                f[8 + i]  = (_Float16)b0[i];
                f[12 + i] = (_Float16)b1[i];
            }
            qfrag[kd] = f;
        }
    }

    // ---------------- streaming-softmax state -------------------------------
    v8f o_acc[8];                      // output, 8 D-slices of 16x16 f32
#pragma unroll
    for (int n = 0; n < 8; ++n) o_acc[n] = (v8f){};
    float m_i[8], l_i[8];              // per C-layout row slot (M = r + 8*half)
#pragma unroll
    for (int r = 0; r < 8; ++r) { m_i[r] = -1e30f; l_i[r] = 0.0f; }

    // KV chunk range for the whole workgroup (union of all waves' windows)
    const int q_wg_lo = qt * QTILE_WG;
    const int q_wg_hi = q_wg_lo + QTILE_WG - 1;
    int cb_lo = q_wg_lo - (WINDOW_ - 1);
    if (cb_lo < 0) cb_lo = 0;
    cb_lo &= ~(KCHUNK - 1);
    const int cb_hi = q_wg_hi & ~(KCHUNK - 1);

    const float* kg = K + (size_t)b * S_ * HKV_ * D_ + (size_t)hkv * D_;
    const float* vg = V + (size_t)b * S_ * HKV_ * D_ + (size_t)hkv * D_;
    const size_t kv_stride = (size_t)HKV_ * D_;   // 1024 floats between kv rows

    // staging decomposition: 32 threads cover one 128-float row as float4s
    const int sd = (tid & 31) * 4;     // d offset of this thread's float4
    const int sr = tid >> 5;           // row offset within group of 4 rows

    _Float16* psw = &ps[wave * 16 * 32];
    const float kInv = SCALE_ / CAP_;

    for (int cb = cb_lo; cb <= cb_hi; cb += KCHUNK) {
        __syncthreads();   // previous chunk's LDS readers done

        // ---- stage K (row-major) and V (transposed) into LDS, f32 -> f16 ----
        // float4 global loads: 512B per wave32 instruction; K stored as packed
        // 4-half ds_store_b64, V scatter-stored transposed (b16 per element).
#pragma unroll
        for (int i = 0; i < KCHUNK / 4; ++i) {
            const int row = i * 4 + sr;
            const size_t gofs = (size_t)(cb + row) * kv_stride + sd;
            v4f kx = *(const v4f*)(kg + gofs);
            v4h kh;
#pragma unroll
            for (int j = 0; j < 4; ++j) kh[j] = (_Float16)kx[j];
            *(v4h*)&kt[row * KT_PITCH + sd] = kh;   // 8B aligned (sd mult of 4)

            v4f vx = *(const v4f*)(vg + gofs);
#pragma unroll
            for (int j = 0; j < 4; ++j)
                vt[(sd + j) * VT_PITCH + row] = (_Float16)vx[j];
        }
        if (cb + KCHUNK <= cb_hi) {   // prefetch next chunk (global_prefetch_b8)
            __builtin_prefetch(kg + (size_t)(cb + KCHUNK + sr) * kv_stride + sd, 0, 1);
            __builtin_prefetch(vg + (size_t)(cb + KCHUNK + sr) * kv_stride + sd, 0, 1);
        }
        __syncthreads();

        // wave-uniform relevance test (must still hit both barriers above)
        const bool active = (cb <= q0 + QTILE_WAVE - 1) &&
                            (cb + KCHUNK - 1 >= q0 - (WINDOW_ - 1));
        if (!active) continue;

        // ---- scores: S = Q * K^T, two 16x16 tiles over the 32 kv columns ----
        v8f c0 = (v8f){}, c1 = (v8f){};
#pragma unroll
        for (int kd = 0; kd < 4; ++kd) {
            // B 16-bit layout: lane col N = l16 (kv row), K element = e<8 ? 8*half+e : 16+8*half+(e-8)
            const v8h* p = (const v8h*)&kt[l16 * KT_PITCH + kd * 32 + 8 * half];
            v16h bk = CAT16(p[0], p[2]);
            c0 = __builtin_amdgcn_wmma_f32_16x16x32_f16(
                     false, qfrag[kd], false, bk, (short)0, c0, false, false);
        }
#pragma unroll
        for (int kd = 0; kd < 4; ++kd) {
            const v8h* p = (const v8h*)&kt[(16 + l16) * KT_PITCH + kd * 32 + 8 * half];
            v16h bk = CAT16(p[0], p[2]);
            c1 = __builtin_amdgcn_wmma_f32_16x16x32_f16(
                     false, qfrag[kd], false, bk, (short)0, c1, false, false);
        }

        // ---- softcap, mask, online softmax (row lives across 16-lane half) --
        float fac[8], p0a[8], p1a[8];
#pragma unroll
        for (int r = 0; r < 8; ++r) {
            const int qi = q0 + r + 8 * half;     // C-layout row M
            const int j0 = cb + l16;              // c0 column (kv index)
            const int j1 = cb + 16 + l16;         // c1 column
            float s0 = CAP_ * tanhf(c0[r] * kInv);
            float s1 = CAP_ * tanhf(c1[r] * kInv);
            const bool v0 = (j0 <= qi) && (qi - j0 < WINDOW_);
            const bool v1 = (j1 <= qi) && (qi - j1 < WINDOW_);
            s0 = v0 ? s0 : -1e30f;
            s1 = v1 ? s1 : -1e30f;

            float mx = fmaxf(s0, s1);             // reduce within 16-lane half
            mx = fmaxf(mx, __shfl_xor(mx, 1, 32));
            mx = fmaxf(mx, __shfl_xor(mx, 2, 32));
            mx = fmaxf(mx, __shfl_xor(mx, 4, 32));
            mx = fmaxf(mx, __shfl_xor(mx, 8, 32));
            const float mnew = fmaxf(m_i[r], mx);

            const float f  = __expf(m_i[r] - mnew);
            const float e0 = (s0 > -1e29f) ? __expf(s0 - mnew) : 0.0f;
            const float e1 = (s1 > -1e29f) ? __expf(s1 - mnew) : 0.0f;
            float rs = e0 + e1;
            rs += __shfl_xor(rs, 1, 32);
            rs += __shfl_xor(rs, 2, 32);
            rs += __shfl_xor(rs, 4, 32);
            rs += __shfl_xor(rs, 8, 32);

            l_i[r] = l_i[r] * f + rs;
            m_i[r] = mnew;
            fac[r] = f;
            p0a[r] = e0;
            p1a[r] = e1;
        }

        // ---- re-layout P (C-layout f32 -> A-layout f16) via per-wave LDS ----
#pragma unroll
        for (int r = 0; r < 8; ++r) {
            const int row = r + 8 * half;
            psw[row * 32 + l16]      = (_Float16)p0a[r];
            psw[row * 32 + 16 + l16] = (_Float16)p1a[r];
        }
        // A layout: lane row = l16; halves at cols [8*half..] and [16+8*half..]
        const v8h* pa0 = (const v8h*)&psw[l16 * 32 + 8 * half];
        const v8h* pa1 = (const v8h*)&psw[l16 * 32 + 16 + 8 * half];
        v16h pafrag = CAT16(pa0[0], pa1[0]);

        // ---- O = diag(fac)*O + P * V : 8 WMMA over D slices -----------------
#pragma unroll
        for (int n = 0; n < 8; ++n) {
#pragma unroll
            for (int r = 0; r < 8; ++r) o_acc[n][r] *= fac[r];
            // B layout: lane col N = l16 -> d = n*16+l16; K element = kv = e<8 ? 8*half+e : 16+8*half+(e-8)
            const v8h* pv = (const v8h*)&vt[(n * 16 + l16) * VT_PITCH + 8 * half];
            v16h bv = CAT16(pv[0], pv[2]);
            o_acc[n] = __builtin_amdgcn_wmma_f32_16x16x32_f16(
                           false, pafrag, false, bv, (short)0, o_acc[n], false, false);
        }
    }

    // ---------------- epilogue: normalize and store --------------------------
#pragma unroll
    for (int r = 0; r < 8; ++r) {
        const int qi = q0 + r + 8 * half;
        const float inv = 1.0f / l_i[r];
        float* row = Out + (((size_t)(b * S_ + qi)) * H_ + h) * D_ + l16;
#pragma unroll
        for (int n = 0; n < 8; ++n) {
            row[n * 16] = o_acc[n][r] * inv;   // lanes 0..15 cover 16 contiguous d
        }
    }
}

extern "C" void kernel_launch(void* const* d_in, const int* in_sizes, int n_in,
                              void* d_out, int out_size, void* d_ws, size_t ws_size,
                              hipStream_t stream) {
    (void)in_sizes; (void)n_in; (void)ws_size; (void)d_ws; (void)out_size;
    const float* q = (const float*)d_in[0];
    const float* k = (const float*)d_in[1];
    const float* v = (const float*)d_in[2];
    float* out = (float*)d_out;

    const int qtiles = S_ / QTILE_WG;              // 32
    dim3 grid(B_ * H_ * qtiles);                   // 2048 workgroups
    dim3 block(128);                               // 4 wave32 waves
    fa_swa_gqa_kernel<<<grid, block, 0, stream>>>(q, k, v, out);
}
